// SimpleRNN_86998857547901
// MI455X (gfx1250) — matrix operand order repaired and checked by reading, not compiled, not run on hardware
//
#include <hip/hip_runtime.h>
#include <hip/hip_bf16.h>

typedef __attribute__((ext_vector_type(16))) __bf16 v16bf;
typedef __attribute__((ext_vector_type(8)))  float  v8f;
typedef int v4i __attribute__((vector_size(16)));

#define N_STEPS 4096
#define N_DATA  8192
#define N_EMB   1024
#define N_H     128
#define N_O     12
#define N_INSZ  1152   // EMB + H
#define N_G     512    // 4*H gate rows

#if __has_builtin(__builtin_amdgcn_global_load_async_to_lds_b128)
#define HAVE_ASYNC_LDS 1
#else
#define HAVE_ASYNC_LDS 0
#endif

// 16-byte copy global -> LDS. Async DMA (ASYNCcnt) when available.
__device__ __forceinline__ void cp16_g2l(const __bf16* __restrict__ g, __bf16* l) {
#if HAVE_ASYNC_LDS
  __builtin_amdgcn_global_load_async_to_lds_b128(
      (__attribute__((address_space(1))) v4i*)(g),
      (__attribute__((address_space(3))) v4i*)(l), 0, 0);
#else
  *(uint4*)l = *(const uint4*)g;
#endif
}

__device__ __forceinline__ void async_wait0() {
#if HAVE_ASYNC_LDS
#if __has_builtin(__builtin_amdgcn_s_wait_asynccnt)
  __builtin_amdgcn_s_wait_asynccnt(0);
#else
  asm volatile("s_wait_asynccnt 0x0" ::: "memory");
#endif
#endif
}

// ---------------------------------------------------------------------------
// Generic f32 -> bf16 conversion (grid-stride)
// ---------------------------------------------------------------------------
__global__ void k_f32_to_bf16(const float* __restrict__ src,
                              __bf16* __restrict__ dst, size_t n) {
  size_t i = (size_t)blockIdx.x * blockDim.x + threadIdx.x;
  size_t stride = (size_t)gridDim.x * blockDim.x;
  for (; i < n; i += stride) dst[i] = (__bf16)src[i];
}

// ---------------------------------------------------------------------------
// Pack one LSTM layer: Wp[512][256] = [Wih (cols 0..inDim-1) | 0 | Whh (cols 128..255)]
// bc[512] = bih + bhh
// ---------------------------------------------------------------------------
__global__ void k_pack_lstm(const float* __restrict__ Wih,
                            const float* __restrict__ Whh,
                            const float* __restrict__ bih,
                            const float* __restrict__ bhh,
                            __bf16* __restrict__ Wp,
                            float* __restrict__ bc, int inDim) {
  int i = blockIdx.x * blockDim.x + threadIdx.x;
  if (i < N_G * 256) {
    int r = i >> 8, c = i & 255;
    float v;
    if (c < 128) v = (c < inDim) ? Wih[r * inDim + c] : 0.0f;
    else         v = Whh[r * N_H + (c - 128)];
    Wp[i] = (__bf16)v;
  }
  if (i < N_G) bc[i] = bih[i] + bhh[i];
}

// Pack W_h2o (12x128) into 16x128 bf16 (rows 12..15 zero) so it is WMMA-tileable.
__global__ void k_pack_h2o(const float* __restrict__ W, __bf16* __restrict__ Wp) {
  int i = blockIdx.x * blockDim.x + threadIdx.x;
  if (i < 16 * N_H) {
    int r = i >> 7, c = i & 127;
    Wp[i] = (__bf16)((r < N_O) ? W[r * N_H + c] : 0.0f);
  }
}

// ---------------------------------------------------------------------------
// emb GEMM: C[4096][1024] = A[4096][8192] * B[1024][8192]^T + bias
// 256 threads (8 waves). Block tile 128x64, BK=64, double-buffered LDS with
// async global->LDS DMA. Wave w owns rows [w*16, w*16+16); A frag reused
// across 4 B subtiles (8 WMMAs per stage per wave).
// ---------------------------------------------------------------------------
__global__ __launch_bounds__(256) void k_emb_gemm(
    const __bf16* __restrict__ A,     // [4096][8192]
    const __bf16* __restrict__ B,     // [1024][8192]  (W_s2i)
    const float*  __restrict__ bias,  // [1024]
    float* __restrict__ Cout)         // [4096][1024]
{
  __shared__ __align__(16) __bf16 As[2][128 * 64];  // 2 x 16 KB
  __shared__ __align__(16) __bf16 Bs[2][64 * 64];   // 2 x  8 KB
  const int tid  = threadIdx.x;
  const int lane = tid & 31, wave = tid >> 5;
  const int hi = lane >> 4, ln = lane & 15;
  const int mBase = blockIdx.y * 128;
  const int nBase = blockIdx.x * 64;

  // Stage one BK=64 slab into buffer `buf`: A 128x64 (1024 chunks), B 64x64 (512).
  auto stage = [&](int buf, int k0) {
#pragma unroll
    for (int r = 0; r < 4; ++r) {          // A: 4 chunks / thread
      int idx = tid + r * 256;
      int row = idx >> 3, kp = (idx & 7) * 8;
      cp16_g2l(&A[(size_t)(mBase + row) * N_DATA + k0 + kp], &As[buf][row * 64 + kp]);
    }
#pragma unroll
    for (int r = 0; r < 2; ++r) {          // B: 2 chunks / thread
      int idx = tid + r * 256;
      int row = idx >> 3, kp = (idx & 7) * 8;
      cp16_g2l(&B[(size_t)(nBase + row) * N_DATA + k0 + kp], &Bs[buf][row * 64 + kp]);
    }
  };

  v8f acc[4] = {};
  stage(0, 0);
  async_wait0();
  __syncthreads();

  const int nKT = N_DATA / 64;  // 128 stages
  for (int kt = 0; kt < nKT; ++kt) {
    int buf = kt & 1;
    if (kt + 1 < nKT) stage(buf ^ 1, (kt + 1) * 64);  // overlap DMA with WMMAs

#pragma unroll
    for (int ks = 0; ks < 64; ks += 32) {
      // A fragment (16-bit A 16x32 layout): lane(l&15)=row, halves split by l>>4.
      v16bf a;
      const __bf16* wr = &As[buf][(wave * 16 + ln) * 64 + ks];
      ((uint4*)&a)[0] = *(const uint4*)(wr + 16 * hi);
      ((uint4*)&a)[1] = *(const uint4*)(wr + 16 * hi + 8);
#pragma unroll
      for (int s = 0; s < 4; ++s) {
        // B fragment (16-bit B 32x16): lane(l&15)=col n, K half by l>>4.
        v16bf b;
        const __bf16* br = &Bs[buf][(s * 16 + ln) * 64 + ks + 16 * hi];
        ((uint4*)&b)[0] = ((const uint4*)br)[0];
        ((uint4*)&b)[1] = ((const uint4*)br)[1];
        acc[s] = __builtin_amdgcn_wmma_f32_16x16x32_bf16(
            false, a, false, b, (short)0, acc[s], false, false);
      }
    }
    async_wait0();
    __syncthreads();
  }

#pragma unroll
  for (int s = 0; s < 4; ++s) {
    int n = nBase + s * 16 + ln;
    float bv = bias[n];
#pragma unroll
    for (int j = 0; j < 8; ++j) {
      int m = mBase + wave * 16 + hi * 8 + j;
      Cout[(size_t)m * N_EMB + n] = acc[s][j] + bv;
    }
  }
}

// ---------------------------------------------------------------------------
// WMMA GEMV helper: out[r] = sum_k W[r][k] * x[k].
// W: global bf16 row-major [rows][cols] (rows %16==0, cols %32==0).
// x broadcast into every B column -> D's rows land in lanes with (lane&15)==0.
// Waves stride over 16-row tiles; full K loop per tile (no cross-wave reduce).
// ---------------------------------------------------------------------------
__device__ __forceinline__ void gemv_wmma(const __bf16* __restrict__ W,
                                          const __bf16* xv, float* out,
                                          int rows, int cols,
                                          int wave, int lane, int nWaves) {
  const int hi = lane >> 4, ln = lane & 15;
  const int nRT = rows >> 4, nKT = cols >> 5;
  for (int rt = wave; rt < nRT; rt += nWaves) {
    v8f acc = {};
    const __bf16* wbase = W + (size_t)(rt * 16 + ln) * cols;
    for (int kt = 0; kt < nKT; ++kt) {
      v16bf a, b;
      const __bf16* wr = wbase + kt * 32;
      ((uint4*)&a)[0] = *(const uint4*)(wr + 16 * hi);
      ((uint4*)&a)[1] = *(const uint4*)(wr + 16 * hi + 8);
      const __bf16* xr = xv + kt * 32 + 16 * hi;
      ((uint4*)&b)[0] = ((const uint4*)xr)[0];
      ((uint4*)&b)[1] = ((const uint4*)xr)[1];
      acc = __builtin_amdgcn_wmma_f32_16x16x32_bf16(
          false, a, false, b, (short)0, acc, false, false);
    }
    if (ln == 0) {
#pragma unroll
      for (int j = 0; j < 8; ++j) out[rt * 16 + hi * 8 + j] = acc[j];
    }
  }
}

// ---------------------------------------------------------------------------
// Persistent sequential scan: 1 block x 512 threads (16 waves) on one WGP.
// All recurrent state lives in LDS; weights stream from L2 (bf16, ~1MB set).
// ---------------------------------------------------------------------------
__global__ __launch_bounds__(512) void k_scan(
    const float*  __restrict__ emb,    // [4096][1024]
    const __bf16* __restrict__ Wl,     // [3][512][256]
    const float*  __restrict__ bc,     // [3][512]
    const __bf16* __restrict__ Wi2h,   // [128][1152]
    const float*  __restrict__ b_i2h,  // [128]
    const __bf16* __restrict__ Wh2h,   // [128][128]
    const float*  __restrict__ b_h2h,  // [128]
    const __bf16* __restrict__ Wh2o,   // [16][128]
    const float*  __restrict__ b_h2o,  // [12]
    const float*  __restrict__ out0,   // [12]
    const float*  __restrict__ h0,     // [3][128]
    const float*  __restrict__ c0,     // [3][128]
    float* __restrict__ outp)          // [4096*12 + 384 + 384]
{
  __shared__ __align__(16) __bf16 sVec[N_INSZ];  // [emb row | x] as bf16
  __shared__ __align__(16) __bf16 sXH[256];      // layer input / hidden vec
  __shared__ float sG[N_G];
  __shared__ float sH[3][N_H], sC[3][N_H];
  __shared__ float sPO[16];
  const int tid  = threadIdx.x;
  const int lane = tid & 31, wave = tid >> 5;

  if (tid < N_O) sPO[tid] = out0[tid];
  else if (tid < 16) sPO[tid] = 0.0f;
  if (tid < N_H) {
    for (int l = 0; l < 3; ++l) {
      sH[l][tid] = h0[l * N_H + tid];
      sC[l][tid] = c0[l * N_H + tid];
    }
  }
  __syncthreads();

  for (int t = 0; t < N_STEPS; ++t) {
    // Stage this step's emb row (f32 -> bf16) and prefetch the next one.
    const float* er = emb + (size_t)t * N_EMB;
    sVec[tid]       = (__bf16)er[tid];
    sVec[tid + 512] = (__bf16)er[tid + 512];
    if (t + 1 < N_STEPS && tid < 64)
      __builtin_prefetch(emb + (size_t)(t + 1) * N_EMB + tid * 16, 0, 0);

    // ---- 3 stacked LSTM layers ----
    for (int l = 0; l < 3; ++l) {
      if (tid < 256) {  // pack [x | h_l] as bf16
        float v;
        if (tid < 128) {
          if (l == 0) v = (tid < N_O) ? sPO[tid] : 0.0f;
          else        v = sH[l - 1][tid];       // fresh h from layer below
        } else {
          v = sH[l][tid - 128];                  // old h of this layer
        }
        sXH[tid] = (__bf16)v;
      }
      __syncthreads();
      gemv_wmma(Wl + (size_t)l * N_G * 256, sXH, sG, N_G, 256, wave, lane, 16);
      __syncthreads();
      if (tid < N_H) {  // gates i,f,g,o stacked in 4*H rows
        const float* b = bc + l * N_G;
        float iv = sG[tid]           + b[tid];
        float fv = sG[N_H + tid]     + b[N_H + tid];
        float gv = sG[2 * N_H + tid] + b[2 * N_H + tid];
        float ov = sG[3 * N_H + tid] + b[3 * N_H + tid];
        iv = 1.0f / (1.0f + __expf(-iv));
        fv = 1.0f / (1.0f + __expf(-fv));
        ov = 1.0f / (1.0f + __expf(-ov));
        gv = tanhf(gv);
        float c2 = fv * sC[l][tid] + iv * gv;
        sC[l][tid] = c2;
        sH[l][tid] = ov * tanhf(c2);
      }
      __syncthreads();
    }

    // ---- FC head: data = [emb_row ; h2] ----
    if (tid < N_H) sVec[N_EMB + tid] = (__bf16)sH[2][tid];
    __syncthreads();
    gemv_wmma(Wi2h, sVec, sG, N_H, N_INSZ, wave, lane, 16);
    __syncthreads();
    if (tid < N_H) sXH[tid] = (__bf16)tanhf(sG[tid] + b_i2h[tid]);
    __syncthreads();
    for (int r = 0; r < 2; ++r) {  // shared-weight h2h applied twice, ReLU
      gemv_wmma(Wh2h, sXH, sG, N_H, N_H, wave, lane, 16);
      __syncthreads();
      if (tid < N_H) {
        float v = sG[tid] + b_h2h[tid];
        sXH[tid] = (__bf16)(v > 0.0f ? v : 0.0f);
      }
      __syncthreads();
    }
    gemv_wmma(Wh2o, sXH, sG, 16, N_H, wave, lane, 16);
    __syncthreads();
    if (tid < N_O) {
      float o = 1.0f / (1.0f + __expf(-(sG[tid] + b_h2o[tid])));
      outp[(size_t)t * N_O + tid] = o;
      sPO[tid] = o;
    }
    __syncthreads();
  }

  if (tid < N_H) {
    for (int l = 0; l < 3; ++l) {
      outp[N_STEPS * N_O + l * N_H + tid]       = sH[l][tid];
      outp[N_STEPS * N_O + 384 + l * N_H + tid] = sC[l][tid];
    }
  }
}

// ---------------------------------------------------------------------------
// Host-side launch
// ---------------------------------------------------------------------------
extern "C" void kernel_launch(void* const* d_in, const int* in_sizes, int n_in,
                              void* d_out, int out_size, void* d_ws, size_t ws_size,
                              hipStream_t stream) {
  const float* inputs  = (const float*)d_in[0];
  const float* output0 = (const float*)d_in[1];
  const float* h0      = (const float*)d_in[2];
  const float* c0      = (const float*)d_in[3];
  const float* W_s2i   = (const float*)d_in[4];
  const float* b_s2i   = (const float*)d_in[5];
  const float* W_i2h   = (const float*)d_in[6];
  const float* b_i2h   = (const float*)d_in[7];
  const float* W_h2h   = (const float*)d_in[8];
  const float* b_h2h   = (const float*)d_in[9];
  const float* W_h2o   = (const float*)d_in[10];
  const float* b_h2o   = (const float*)d_in[11];
  // Per-layer LSTM params at 12 + 4*l
  const float* Wih[3] = {(const float*)d_in[12], (const float*)d_in[16], (const float*)d_in[20]};
  const float* Whh[3] = {(const float*)d_in[13], (const float*)d_in[17], (const float*)d_in[21]};
  const float* bih[3] = {(const float*)d_in[14], (const float*)d_in[18], (const float*)d_in[22]};
  const float* bhh[3] = {(const float*)d_in[15], (const float*)d_in[19], (const float*)d_in[23]};
  const int inDims[3] = {N_O, N_H, N_H};

  uint8_t* ws = (uint8_t*)d_ws;
  size_t off = 0;
  auto carve = [&](size_t bytes) -> void* {
    void* p = ws + off;
    off = (off + bytes + 255) & ~(size_t)255;
    return p;
  };
  __bf16* inA_bf   = (__bf16*)carve((size_t)N_STEPS * N_DATA * 2);   // 64 MB
  __bf16* Ws2i_bf  = (__bf16*)carve((size_t)N_EMB * N_DATA * 2);     // 16 MB
  float*  embf     = (float*) carve((size_t)N_STEPS * N_EMB * 4);    // 16 MB
  __bf16* Wl_bf    = (__bf16*)carve((size_t)3 * N_G * 256 * 2);      // 768 KB
  float*  bc_f     = (float*) carve((size_t)3 * N_G * 4);
  __bf16* Wi2h_bf  = (__bf16*)carve((size_t)N_H * N_INSZ * 2);
  __bf16* Wh2h_bf  = (__bf16*)carve((size_t)N_H * N_H * 2);
  __bf16* Wh2o_bf  = (__bf16*)carve((size_t)16 * N_H * 2);

  // 1) Convert big operands to bf16
  k_f32_to_bf16<<<4096, 256, 0, stream>>>(inputs, inA_bf, (size_t)N_STEPS * N_DATA);
  k_f32_to_bf16<<<2048, 256, 0, stream>>>(W_s2i, Ws2i_bf, (size_t)N_EMB * N_DATA);
  k_f32_to_bf16<<<576, 256, 0, stream>>>(W_i2h, Wi2h_bf, (size_t)N_H * N_INSZ);
  k_f32_to_bf16<<<64, 256, 0, stream>>>(W_h2h, Wh2h_bf, (size_t)N_H * N_H);
  k_pack_h2o<<<8, 256, 0, stream>>>(W_h2o, Wh2o_bf);

  // 2) Pack LSTM layers
  for (int l = 0; l < 3; ++l) {
    k_pack_lstm<<<(N_G * 256 + 255) / 256, 256, 0, stream>>>(
        Wih[l], Whh[l], bih[l], bhh[l],
        Wl_bf + (size_t)l * N_G * 256, bc_f + (size_t)l * N_G, inDims[l]);
  }

  // 3) Batched s2i GEMM via WMMA (async double-buffered LDS staging)
  dim3 ggrid(N_EMB / 64, N_STEPS / 128);
  k_emb_gemm<<<ggrid, 256, 0, stream>>>(inA_bf, Ws2i_bf, b_s2i, embf);

  // 4) Persistent sequential scan (single WGP, WMMA GEMVs)
  k_scan<<<1, 512, 0, stream>>>(embf, Wl_bf, bc_f, Wi2h_bf, b_i2h,
                                Wh2h_bf, b_h2h, Wh2o_bf, b_h2o,
                                output0, h0, c0, (float*)d_out);
}
